// XModel_66795331387584
// MI455X (gfx1250) — matrix operand order repaired and checked by reading, not compiled, not run on hardware
//
#include <hip/hip_runtime.h>
#include <hip/hip_bf16.h>

typedef __attribute__((ext_vector_type(16))) _Float16 v16h;
typedef __attribute__((ext_vector_type(8)))  _Float16 v8h;
typedef __attribute__((ext_vector_type(8)))  float    v8f;

// LDS tile layout: 32 rows x 16 floats. Rows 0-15 at r*64B, rows 16-31 shifted
// +64B,
// while every 16B chunk stays 16B-aligned for async b128 writes.
#define ROWOFF(r) (((r) << 4) + ((r) & 16))   // in floats
#define TILE_FLOATS 528                        // ROWOFF(31)+16 = 528

static __device__ inline v8f wmma_f16(v16h a, v16h b, v8f c) {
  // D = A(16x32 f16) * B(32x16 f16) + C(16x16 f32)
  return __builtin_amdgcn_wmma_f32_16x16x32_f16(
      /*neg_a=*/false, a, /*neg_b=*/false, b,
      /*c_mod=*/(short)0, c, /*reuse_a=*/false, /*reuse_b=*/false);
}

// ---- async staging: 32x16 f32 tile, one b128 chunk per lane (128 threads) --
static __device__ inline void async_stage_tile(float* tile, const float* __restrict__ src,
                                               int ld, int j0, int c0, int nrows, int ncols) {
  int idx = threadIdx.x & 127;
  int r = idx >> 2, c = idx & 3;
  int j = j0 + r;
  if (j < nrows && (c0 + (c + 1) * 4) <= ncols) {
    unsigned ldsb = (unsigned)(uintptr_t)(tile + ROWOFF(r) + c * 4);
    unsigned long long ga =
        (unsigned long long)(uintptr_t)(src + (long)j * ld + c0 + c * 4);
    asm volatile("global_load_async_to_lds_b128 %0, %1, off"
                 :: "v"(ldsb), "v"(ga) : "memory");
  }
}

// ---- async staging of the two 32x8 "treatment" row blocks (M and X) -------
// waves 0,1 move M; waves 2,3 move X -> exactly one async instr per wave.
static __device__ inline void async_stage_mx(float* ldsM, float* ldsX,
                                             const float* __restrict__ Msrc,
                                             const float* __restrict__ Xsrc,
                                             int j0, int nrows) {
  int idx = threadIdx.x & 127;
  const float* src = (idx < 64) ? Msrc : Xsrc;   // uniform per wave
  float*       dst = (idx < 64) ? ldsM : ldsX;
  int t = idx & 63;
  int r = t >> 1, c = t & 1;
  int j = j0 + r;
  if (j < nrows) {
    unsigned ldsb = (unsigned)(uintptr_t)(dst + r * 8 + c * 4);
    unsigned long long ga =
        (unsigned long long)(uintptr_t)(src + (long)j * 8 + c * 4);
    asm volatile("global_load_async_to_lds_b128 %0, %1, off"
                 :: "v"(ldsb), "v"(ga) : "memory");
  }
}

static __device__ inline void wait_async4(bool depth4) {
  if (depth4) asm volatile("s_wait_asynccnt 0x4" ::: "memory");
  else        asm volatile("s_wait_asynccnt 0x0" ::: "memory");
}
static __device__ inline void wait_async1(bool depth1) {
  if (depth1) asm volatile("s_wait_asynccnt 0x1" ::: "memory");
  else        asm volatile("s_wait_asynccnt 0x0" ::: "memory");
}

// ---- DS-store zeros into rows >= rvalid (disjoint from async writes) ------
static __device__ inline void zero_tail5(float* t0, float* t1, float* t2,
                                         float* m0, float* m1, int rvalid) {
  for (int idx = threadIdx.x; idx < 32 * 16; idx += blockDim.x) {
    int r = idx >> 4, c = idx & 15;
    if (r >= rvalid) {
      int o = ROWOFF(r) + c;
      t0[o] = 0.f; t1[o] = 0.f; t2[o] = 0.f;
    }
  }
  for (int idx = threadIdx.x; idx < 32 * 8; idx += blockDim.x) {
    if ((idx >> 3) >= rvalid) { m0[idx] = 0.f; m1[idx] = 0.f; }
  }
}
static __device__ inline void zero_tail1(float* t0, int rvalid) {
  for (int idx = threadIdx.x; idx < 32 * 16; idx += blockDim.x) {
    int r = idx >> 4, c = idx & 15;
    if (r >= rvalid) t0[ROWOFF(r) + c] = 0.f;
  }
}

// ---- A-fragment (16x32 f16) from an f16 matrix, tail-safe -----------------
static __device__ inline v16h load_a_frag_f16(const _Float16* __restrict__ src,
                                              int ld, int r0, int j0, int rows, int K) {
  int lane = threadIdx.x & 31;
  int row = r0 + (lane & 15); if (row >= rows) row = rows - 1;
  int kb = (lane < 16) ? 0 : 8;
  const _Float16* p = src + (long)row * ld + j0 + kb;
  v8h lo = *(const v8h*)p;
  v8h hi;
  if (j0 + 32 <= K) {
    hi = *(const v8h*)(p + 16);
  } else {
#pragma unroll
    for (int i = 0; i < 8; ++i) hi[i] = (_Float16)0.f;
  }
  v16h a;
#pragma unroll
  for (int i = 0; i < 8; ++i) { a[i] = lo[i]; a[8 + i] = hi[i]; }
  return a;
}

// ---- A-fragment from an f32 matrix with on-the-fly f16 convert ------------
static __device__ inline v16h load_a_frag_f32(const float* __restrict__ src,
                                              int ld, int r0, int j0, int rows, int K) {
  int lane = threadIdx.x & 31;
  int row = r0 + (lane & 15); if (row >= rows) row = rows - 1;
  int kb = (lane < 16) ? 0 : 8;
  const float* p = src + (long)row * ld + j0 + kb;
  v16h a;
  bool tail = (j0 + 32 > K);
#pragma unroll
  for (int i = 0; i < 8; ++i) a[i] = (_Float16)p[i];
  if (!tail) {
#pragma unroll
    for (int i = 0; i < 8; ++i) a[8 + i] = (_Float16)p[16 + i];
  } else {
#pragma unroll
    for (int i = 0; i < 8; ++i) a[8 + i] = (_Float16)0.f;
  }
  return a;
}

// ===========================================================================
// Phase features: P = alpha @ W^T, emit cos(P), sin(P) as f16.
// ===========================================================================
__global__ void phase_kernel(const float* __restrict__ alpha, const float* __restrict__ W,
                             _Float16* __restrict__ cosP, _Float16* __restrict__ sinP,
                             int a, int n) {
  long idx = (long)blockIdx.x * blockDim.x + threadIdx.x;
  long total = (long)a * n;
  if (idx >= total) return;
  int ai = (int)(idx / n), ji = (int)(idx % n);
  float s = 0.f;
#pragma unroll
  for (int d = 0; d < 8; ++d) s += alpha[(long)ai * 8 + d] * W[(long)ji * 8 + d];
  float sn, cs;
  __sincosf(s, &sn, &cs);
  cosP[idx] = (_Float16)cs;
  sinP[idx] = (_Float16)sn;
}

// ===========================================================================
// C = A @ B  (f32 storage, f16 WMMA with f32 accumulate), async-staged and
// double-buffered B tile shared by 4 waves; one 16x16 C tile per wave.
// ===========================================================================
__global__ __launch_bounds__(128) void gemm_wmma_kernel(
    const float* __restrict__ A, const float* __restrict__ B, float* __restrict__ C,
    int Mn, int Nn, int Kn) {
  __shared__ float ldsB[2][TILE_FLOATS];
  int wid = threadIdx.x >> 5, lane = threadIdx.x & 31;
  int zc = lane & 15;
  int rbase = (lane < 16) ? 0 : 8;   // K base for B fragment
  int abase = (lane < 16) ? 0 : 8;   // M base for C layout
  int r0 = (blockIdx.y * 4 + wid) * 16;
  int c0 = blockIdx.x * 16;
  int niters = (Kn + 31) / 32;

  // prologue: stage tile 0
  async_stage_tile(ldsB[0], B, Nn, 0, c0, Kn, Nn);
  if (Kn < 32) zero_tail1(ldsB[0], Kn);

  v8f acc = {};
  for (int jt = 0; jt < niters; ++jt) {
    int cur = jt & 1;
    int j0 = jt * 32;
    bool more = (jt + 1) < niters;
    if (more) {
      async_stage_tile(ldsB[cur ^ 1], B, Nn, j0 + 32, c0, Kn, Nn);
      int rv = Kn - (j0 + 32);
      if (rv < 32) zero_tail1(ldsB[cur ^ 1], rv);
    }
    // async loads complete in order: depth-4? no -> depth = #just issued (1),
    // unless the issued tile was partial (whole waves may have skipped issue).
    wait_async1(more && (j0 + 64 <= Kn));
    __syncthreads();

    v16h af = load_a_frag_f32(A, Kn, r0, j0, Mn, Kn);
    v16h bf;
#pragma unroll
    for (int h = 0; h < 16; ++h) bf[h] = (_Float16)ldsB[cur][ROWOFF(rbase + h + (h & 8)) + zc];
    acc = wmma_f16(af, bf, acc);
    __syncthreads();
  }
#pragma unroll
  for (int v = 0; v < 8; ++v) {
    int r = r0 + v + abase;
    if (r < Mn && (c0 + zc) < Nn) C[(long)r * Nn + c0 + zc] = acc[v];
  }
}

// ===========================================================================
// G = K_Z1Z2 - T1/c + T2/c^2   (c = n * exp(log_lambda); G = c * gamma_X —
// the 1/c prefactor cancels in num/den, and keeps G in f16-friendly range)
// ===========================================================================
__global__ void combine_kernel(const float* __restrict__ K12, const float* __restrict__ T1,
                               const float* __restrict__ T2, const float* __restrict__ loglam,
                               float* __restrict__ G, int n, long total) {
  long i = (long)blockIdx.x * blockDim.x + threadIdx.x;
  if (i >= total) return;
  float c = (float)n * __expf(loglam[0]);
  float ic = 1.f / c;
  G[i] = K12[i] - T1[i] * ic + T2[i] * (ic * ic);
}

__global__ void init_out_kernel(float* out) {
  if (threadIdx.x == 0 && blockIdx.x == 0) out[0] = 0.f;
}

// ===========================================================================
// Fused labels + loss. Per 16(a)x16(z) tile accumulate den_MN, den_X and
// num_MN[d], num_X[d] (d split into two passes of 4 to cap VGPR pressure),
// then epilogue |num_MN/den_MN - num_X/den_X|^2 -> one atomicAdd per wave.
// All gamma/M/X staging goes through async-to-LDS with double buffering.
// ===========================================================================
__global__ __launch_bounds__(128) void fused_loss_kernel(
    const _Float16* __restrict__ cosN, const _Float16* __restrict__ sinN,
    const _Float16* __restrict__ cosX, const _Float16* __restrict__ sinX,
    const float* __restrict__ Mrows, const float* __restrict__ Xrows,
    const float* __restrict__ gMN, const float* __restrict__ gN,
    const float* __restrict__ G, float* __restrict__ out,
    int a, int n, int z) {
  __shared__ float ldsMN[2][TILE_FLOATS];
  __shared__ float ldsNN[2][TILE_FLOATS];
  __shared__ float ldsGG[2][TILE_FLOATS];
  __shared__ float ldsM [2][32 * 8];
  __shared__ float ldsX [2][32 * 8];

  int wid = threadIdx.x >> 5, lane = threadIdx.x & 31;
  int zc = lane & 15;
  int rbase = (lane < 16) ? 0 : 8;   // K base for B fragments
  int abase = (lane < 16) ? 0 : 8;   // M base for C layout
  int a0 = (blockIdx.y * 4 + wid) * 16;
  int z0 = blockIdx.x * 16;
  int niters = (n + 31) / 32;

  v8f zero = {};
  v8f dMNr = zero, dMNi = zero, dXr = zero, dXi = zero;  // live across both passes
  v8f loss = zero;

  for (int pass = 0; pass < 2; ++pass) {
    v8f numr[4], numi[4], xnr[4], xni[4];
#pragma unroll
    for (int t = 0; t < 4; ++t) { numr[t] = zero; numi[t] = zero; xnr[t] = zero; xni[t] = zero; }
    int dbase = pass * 4;

    // stage tile 0 of this pass (prev pass ended with a barrier -> WAR safe)
    async_stage_tile(ldsMN[0], gMN, z, 0, z0, n, z);
    async_stage_tile(ldsNN[0], gN,  z, 0, z0, n, z);
    async_stage_tile(ldsGG[0], G,   z, 0, z0, n, z);
    async_stage_mx(ldsM[0], ldsX[0], Mrows, Xrows, 0, n);
    if (n < 32) zero_tail5(ldsMN[0], ldsNN[0], ldsGG[0], ldsM[0], ldsX[0], n);

    for (int jt = 0; jt < niters; ++jt) {
      int cur = jt & 1;
      int j0 = jt * 32;
      bool more = (jt + 1) < niters;
      if (more) {
        int nj = j0 + 32;
        async_stage_tile(ldsMN[cur ^ 1], gMN, z, nj, z0, n, z);
        async_stage_tile(ldsNN[cur ^ 1], gN,  z, nj, z0, n, z);
        async_stage_tile(ldsGG[cur ^ 1], G,   z, nj, z0, n, z);
        async_stage_mx(ldsM[cur ^ 1], ldsX[cur ^ 1], Mrows, Xrows, nj, n);
        int rv = n - nj;
        if (rv < 32) zero_tail5(ldsMN[cur ^ 1], ldsNN[cur ^ 1], ldsGG[cur ^ 1],
                                ldsM[cur ^ 1], ldsX[cur ^ 1], rv);
      }
      // in-order async completion: wait to depth of the just-issued batch (4);
      // fall back to 0 when the issued tile is partial (issue may be skipped).
      wait_async4(more && (j0 + 64 <= n));
      __syncthreads();

      v16h acN = load_a_frag_f16(cosN, n, a0, j0, a, n);
      v16h asN = load_a_frag_f16(sinN, n, a0, j0, a, n);
      v16h acX = load_a_frag_f16(cosX, n, a0, j0, a, n);
      v16h asX = load_a_frag_f16(sinX, n, a0, j0, a, n);

      // hoist this lane's gamma column once; reused by all d fragments
      float gmn[16], gg[16];
#pragma unroll
      for (int h = 0; h < 16; ++h) gmn[h] = ldsMN[cur][ROWOFF(rbase + h + (h & 8)) + zc];
#pragma unroll
      for (int h = 0; h < 16; ++h) gg[h]  = ldsGG[cur][ROWOFF(rbase + h + (h & 8)) + zc];

      if (pass == 0) {  // denominators only need one sweep
        v16h bN, bG;
#pragma unroll
        for (int h = 0; h < 16; ++h) bN[h] = (_Float16)ldsNN[cur][ROWOFF(rbase + h + (h & 8)) + zc];
#pragma unroll
        for (int h = 0; h < 16; ++h) bG[h] = (_Float16)gg[h];
        dMNr = wmma_f16(acN, bN, dMNr);
        dMNi = wmma_f16(asN, bN, dMNi);
        dXr  = wmma_f16(acX, bG, dXr);
        dXi  = wmma_f16(asX, bG, dXi);
      }

#pragma unroll
      for (int dd = 0; dd < 4; ++dd) {
        int d = dbase + dd;
        v16h bMN, bX;
#pragma unroll
        for (int h = 0; h < 16; ++h)
          bMN[h] = (_Float16)(gmn[h] * ldsM[cur][(rbase + h + (h & 8)) * 8 + d]);
#pragma unroll
        for (int h = 0; h < 16; ++h)
          bX[h]  = (_Float16)(gg[h]  * ldsX[cur][(rbase + h + (h & 8)) * 8 + d]);
        numr[dd] = wmma_f16(acN, bMN, numr[dd]);
        numi[dd] = wmma_f16(asN, bMN, numi[dd]);
        xnr[dd]  = wmma_f16(acX, bX,  xnr[dd]);
        xni[dd]  = wmma_f16(asX, bX,  xni[dd]);
      }
      __syncthreads();  // protect WAR on cur buffer before next issue into it
    }

    // epilogue for this half of d: label = num * conj(den) / |den|^2
#pragma unroll
    for (int v = 0; v < 8; ++v) {
      float idMN = 1.f / (dMNr[v] * dMNr[v] + dMNi[v] * dMNi[v]);
      float idX  = 1.f / (dXr[v] * dXr[v] + dXi[v] * dXi[v]);
#pragma unroll
      for (int dd = 0; dd < 4; ++dd) {
        float lMNr = (numr[dd][v] * dMNr[v] + numi[dd][v] * dMNi[v]) * idMN;
        float lMNi = (numi[dd][v] * dMNr[v] - numr[dd][v] * dMNi[v]) * idMN;
        float lXr  = (xnr[dd][v] * dXr[v] + xni[dd][v] * dXi[v]) * idX;
        float lXi  = (xni[dd][v] * dXr[v] - xnr[dd][v] * dXi[v]) * idX;
        float dr = lMNr - lXr, di = lMNi - lXi;
        loss[v] += dr * dr + di * di;
      }
    }
  }

  // mask invalid rows, wave-reduce, single atomic per wave
  float part = 0.f;
  if (z0 + zc < z) {
#pragma unroll
    for (int v = 0; v < 8; ++v) {
      int ar = a0 + v + abase;
      if (ar < a) part += loss[v];
    }
  }
  part *= 1.f / ((float)a * (float)z);
#pragma unroll
  for (int off = 16; off > 0; off >>= 1) part += __shfl_xor(part, off, 32);
  if (lane == 0) atomicAdd(out, part);
}

// ===========================================================================
extern "C" void kernel_launch(void* const* d_in, const int* in_sizes, int n_in,
                              void* d_out, int out_size, void* d_ws, size_t ws_size,
                              hipStream_t stream) {
  const float* M      = (const float*)d_in[0];
  const float* N      = (const float*)d_in[1];
  const float* X      = (const float*)d_in[2];
  const float* loglam = (const float*)d_in[3];
  const float* K11    = (const float*)d_in[4];
  const float* K12    = (const float*)d_in[5];
  const float* gMN    = (const float*)d_in[6];
  const float* gN     = (const float*)d_in[7];
  const float* alpha  = (const float*)d_in[8];

  const int D = 8;
  int n = in_sizes[0] / D;      // 2000
  int a = in_sizes[8] / D;      // 1000
  int z = in_sizes[5] / n;      // 2000

  // workspace layout (16B aligned): T1,T2,G f32 [n,z]; cos/sin f16 [a,n]
  char* ws = (char*)d_ws;
  size_t fsz = (size_t)n * z * sizeof(float);
  size_t hsz = (size_t)a * n * sizeof(_Float16);
  float*    T1   = (float*)(ws);
  float*    T2   = (float*)(ws + fsz);
  float*    G    = (float*)(ws + 2 * fsz);
  _Float16* cosN = (_Float16*)(ws + 3 * fsz);
  _Float16* sinN = (_Float16*)(ws + 3 * fsz + hsz);
  _Float16* cosX = (_Float16*)(ws + 3 * fsz + 2 * hsz);
  _Float16* sinX = (_Float16*)(ws + 3 * fsz + 3 * hsz);

  init_out_kernel<<<1, 32, 0, stream>>>((float*)d_out);

  long pt = (long)a * n;
  int  pb = (int)((pt + 255) / 256);
  phase_kernel<<<pb, 256, 0, stream>>>(alpha, N, cosN, sinN, a, n);
  phase_kernel<<<pb, 256, 0, stream>>>(alpha, X, cosX, sinX, a, n);

  // Neumann-series ridge solve: cond(K + cI) ~ 1.002 so 3 terms suffice.
  dim3 gg((z + 15) / 16, (n + 63) / 64);
  gemm_wmma_kernel<<<gg, 128, 0, stream>>>(K11, K12, T1, n, z, n);
  gemm_wmma_kernel<<<gg, 128, 0, stream>>>(K11, T1, T2, n, z, n);

  long tot = (long)n * z;
  combine_kernel<<<(int)((tot + 255) / 256), 256, 0, stream>>>(K12, T1, T2, loglam, G, n, tot);

  dim3 gf((z + 15) / 16, (a + 63) / 64);
  fused_loss_kernel<<<gf, 128, 0, stream>>>(cosN, sinN, cosX, sinX, M, X, gMN, gN, G,
                                            (float*)d_out, a, n, z);
}
